// SingleAttention_25967372271707
// MI455X (gfx1250) — matrix operand order repaired and checked by reading, not compiled, not run on hardware
//
#include <hip/hip_runtime.h>
#include <hip/hip_bf16.h>

#define BB 8
#define SS 2048
#define FF 256
#define EE 64
#define DKK 256
#define IN_DIM (FF + 2 * EE)  // 384

typedef _Float16 half8  __attribute__((ext_vector_type(8)));
typedef _Float16 half16 __attribute__((ext_vector_type(16)));
typedef float    float8 __attribute__((ext_vector_type(8)));

__device__ __forceinline__ half16 cat8(half8 lo, half8 hi) {
  return __builtin_shufflevector(lo, hi, 0,1,2,3,4,5,6,7,8,9,10,11,12,13,14,15);
}

// ---------------------------------------------------------------------------
// Kernel 1: Q/K projection. q = concat(seq,pos,time) @ Wq^T (1/sqrt(DK)=1/16
// folded into Qh), k unscaled. Output f16. One block per (b,s) row.
// ---------------------------------------------------------------------------
__global__ __launch_bounds__(256)
void proj_kernel(const float* __restrict__ seq, const float* __restrict__ pos,
                 const float* __restrict__ timee, const float* __restrict__ Wq,
                 const float* __restrict__ Wk, _Float16* __restrict__ Qh,
                 _Float16* __restrict__ Kh)
{
  __shared__ __align__(16) float row[IN_DIM];
  const int r = blockIdx.x;            // b*S + s
  const int t = threadIdx.x;
  row[t] = seq[(size_t)r * FF + t];
  if (t < EE) {
    row[FF + t]      = pos[(size_t)r * EE + t];
    row[FF + EE + t] = timee[(size_t)r * EE + t];
  }
  __syncthreads();
  const float4* wq = (const float4*)(Wq + (size_t)t * IN_DIM);
  const float4* wk = (const float4*)(Wk + (size_t)t * IN_DIM);
  const float4* rw = (const float4*)row;
  float accq = 0.f, acck = 0.f;
#pragma unroll 4
  for (int i = 0; i < IN_DIM / 4; ++i) {
    float4 a  = rw[i];
    float4 q4 = wq[i];
    float4 k4 = wk[i];
    accq += a.x * q4.x + a.y * q4.y + a.z * q4.z + a.w * q4.w;
    acck += a.x * k4.x + a.y * k4.y + a.z * k4.z + a.w * k4.w;
  }
  Qh[(size_t)r * DKK + t] = (_Float16)(accq * 0.0625f);
  Kh[(size_t)r * DKK + t] = (_Float16)acck;
}

// ---------------------------------------------------------------------------
// Kernel 2: V^T staging. Vt[b][f][s] = (f16) seq[b][s][f] via 16x16 LDS tiles.
// ---------------------------------------------------------------------------
__global__ __launch_bounds__(256)
void transpose_kernel(const float* __restrict__ seq, _Float16* __restrict__ Vt)
{
  __shared__ _Float16 tile[16][17];
  const int nft = FF / 16;                      // 16
  const int bt  = blockIdx.x;
  const int b   = bt / ((SS / 16) * nft);
  const int rr  = bt % ((SS / 16) * nft);
  const int s0  = (rr / nft) * 16;
  const int f0  = (rr % nft) * 16;
  const int tx  = threadIdx.x & 15;
  const int ty  = threadIdx.x >> 4;
  tile[ty][tx] = (_Float16)seq[((size_t)(b * SS + s0 + ty)) * FF + f0 + tx];
  __syncthreads();
  Vt[((size_t)(b * FF + f0 + ty)) * SS + s0 + tx] = tile[tx][ty];
}

// ---------------------------------------------------------------------------
// Kernel 3: flash attention. One wave per 16-query tile, 32 keys/iteration.
// Double-buffered K/V fragment loads, WMMA reuse_a hints on repeated-A ops,
// mask prefetch for the next key tile.
// ---------------------------------------------------------------------------
__global__ __launch_bounds__(128)
void attn_kernel(const _Float16* __restrict__ Qh, const _Float16* __restrict__ Kh,
                 const _Float16* __restrict__ Vt, const float* __restrict__ mask,
                 const int* __restrict__ causality, float* __restrict__ out)
{
  // 40-half row pitch (80B): lanes 0..15 start on all multiples of 4 banks ->
  // conflict-free ds_load_b128 of the A fragment.
  __shared__ __align__(16) _Float16 Plds[4][16][40];

  const int wave = threadIdx.x >> 5;
  const int lane = threadIdx.x & 31;
  const int hf   = lane >> 4;                  // half-wave: 0 or 1
  const int ln   = lane & 15;
  const int tile = blockIdx.x * 4 + wave;      // 0 .. B*S/16 - 1
  const int b    = tile >> 7;                  // tile / (S/16)
  const int qi0  = (tile & 127) << 4;
  const int causal = causality[0];

  // ---- Q A-fragments: lane ln -> row qi0+ln; halves h -> k=16*hf+h
  half16 qfrag[8];
  const _Float16* qrow = Qh + ((size_t)(b * SS + qi0 + ln)) * DKK;
#pragma unroll
  for (int c = 0; c < 8; ++c) {
    const int k0 = c * 32 + hf * 16;
    half8 lo = *(const half8*)(qrow + k0);
    half8 hi = *(const half8*)(qrow + k0 + 8);
    qfrag[c] = cat8(lo, hi);
  }

  float8 o[16];
#pragma unroll
  for (int c = 0; c < 16; ++c) o[c] = (float8){0.f,0.f,0.f,0.f,0.f,0.f,0.f,0.f};
  float mrun[8], lrun[8];
#pragma unroll
  for (int v = 0; v < 8; ++v) { mrun[v] = -1e30f; lrun[v] = 0.f; }

  const int j0start = causal ? (qi0 & ~31) : 0;
  for (int j0 = j0start; j0 < SS; j0 += 32) {
    // ---- scores: two 16x16 tiles over 32 keys, K-dim 256 = 8 x 32 chunks.
    // B operand: lane ln -> key j0+ln; halves h -> kd = c*32 + 16*hf + h.
    float8 s0 = (float8){0.f,0.f,0.f,0.f,0.f,0.f,0.f,0.f};
    float8 s1 = (float8){0.f,0.f,0.f,0.f,0.f,0.f,0.f,0.f};
    const _Float16* krow0 = Kh + ((size_t)(b * SS + j0 + ln)) * DKK + hf * 16;
    const _Float16* krow1 = krow0 + 16 * DKK;
    // double-buffered fragment rotation
    half16 kb0 = *(const half16*)(krow0);
    half16 kb1 = *(const half16*)(krow1);
#pragma unroll
    for (int c = 0; c < 8; ++c) {
      half16 nb0 = kb0, nb1 = kb1;
      if (c < 7) {
        nb0 = *(const half16*)(krow0 + (c + 1) * 32);
        nb1 = *(const half16*)(krow1 + (c + 1) * 32);
      }
      s0 = __builtin_amdgcn_wmma_f32_16x16x32_f16(false, qfrag[c], false, kb0,
                                                  (short)0, s0, false, false);
      // identical op, same A register -> RA reuse hint is legal (ISA 7.12)
      s1 = __builtin_amdgcn_wmma_f32_16x16x32_f16(false, qfrag[c], false, kb1,
                                                  (short)0, s1, true, false);
      kb0 = nb0;
      kb1 = nb1;
    }
    if (j0 + 32 < SS) __builtin_prefetch(krow0 + 32 * DKK, 0, 1);  // next K tile

    // ---- clamp(10) -> causal(j>=i) & cluster mask -> online softmax
    const float* mrow = mask + ((size_t)(b * SS + qi0 + hf * 8)) * SS + j0 + ln;
    float alpha[8];
#pragma unroll
    for (int v = 0; v < 8; ++v) {
      const int i  = qi0 + v + hf * 8;  // C layout: m = v + 8*hf, n = ln
      const int jA = j0 + ln;
      const int jB = jA + 16;
      float a0 = fminf(s0[v], 10.f);
      float a1 = fminf(s1[v], 10.f);
      const float mk0 = mrow[(size_t)v * SS];
      const float mk1 = mrow[(size_t)v * SS + 16];
      if (j0 + 32 < SS) __builtin_prefetch(mrow + (size_t)v * SS + 32, 0, 1);
      const bool okA = (mk0 > 0.f) && (!causal || (jA >= i));
      const bool okB = (mk1 > 0.f) && (!causal || (jB >= i));
      a0 = okA ? a0 : -1e30f;
      a1 = okB ? a1 : -1e30f;
      // row-max over the 32 key columns (16 lanes of this half-wave x 2 tiles)
      float t = fmaxf(a0, a1);
      t = fmaxf(t, __shfl_xor(t, 1, 32));
      t = fmaxf(t, __shfl_xor(t, 2, 32));
      t = fmaxf(t, __shfl_xor(t, 4, 32));
      t = fmaxf(t, __shfl_xor(t, 8, 32));
      const float mnew = fmaxf(mrun[v], t);
      alpha[v] = __expf(mrun[v] - mnew);
      mrun[v]  = mnew;
      const float p0 = __expf(a0 - mnew);
      const float p1 = __expf(a1 - mnew);
      float rs = p0 + p1;
      rs += __shfl_xor(rs, 1, 32);
      rs += __shfl_xor(rs, 2, 32);
      rs += __shfl_xor(rs, 4, 32);
      rs += __shfl_xor(rs, 8, 32);
      lrun[v] = lrun[v] * alpha[v] + rs;
      const int m = v + hf * 8;
      Plds[wave][m][ln]      = (_Float16)p0;
      Plds[wave][m][ln + 16] = (_Float16)p1;
    }
    // rescale O accumulators by per-row alpha (co-executes with WMMA/TRANS)
#pragma unroll
    for (int c = 0; c < 16; ++c)
#pragma unroll
      for (int v = 0; v < 8; ++v) o[c][v] *= alpha[v];

    asm volatile("s_wait_dscnt 0" ::: "memory");  // wave-private LDS RAW

    // ---- P (16x32) as A operand from LDS
    half16 pfrag;
    {
      const _Float16* prow = &Plds[wave][ln][0];
      const int k0 = hf * 16;
      half8 lo = *(const half8*)(prow + k0);
      half8 hi = *(const half8*)(prow + k0 + 8);
      pfrag = cat8(lo, hi);
    }

    // ---- O += P @ V : B operand from Vt, lane ln -> col c*16+ln,
    //      halves h -> key j0 + 16*hf + h (contiguous along s in Vt).
    //      Double-buffered; A (pfrag) constant -> RA reuse on c>0
    //      (first WMMA peeled so the reuse flag is a literal constant).
    const _Float16* vbase = Vt + ((size_t)(b * FF + ln)) * SS + j0 + hf * 16;
    half16 vb = *(const half16*)(vbase);
    {
      half16 vn = *(const half16*)(vbase + (size_t)16 * SS);
      o[0] = __builtin_amdgcn_wmma_f32_16x16x32_f16(false, pfrag, false, vb,
                                                    (short)0, o[0], false, false);
      vb = vn;
    }
#pragma unroll
    for (int c = 1; c < 16; ++c) {
      half16 vn = vb;
      if (c < 15) vn = *(const half16*)(vbase + (size_t)(c + 1) * 16 * SS);
      o[c] = __builtin_amdgcn_wmma_f32_16x16x32_f16(false, pfrag, false, vb,
                                                    (short)0, o[c], true, false);
      vb = vn;
    }
  }

  // ---- epilogue: O / l, scatter per C layout
  float inv[8];
#pragma unroll
  for (int v = 0; v < 8; ++v) inv[v] = 1.f / lrun[v];
#pragma unroll
  for (int c = 0; c < 16; ++c)
#pragma unroll
    for (int v = 0; v < 8; ++v)
      out[((size_t)(b * SS + qi0 + v + hf * 8)) * FF + c * 16 + ln] =
          o[c][v] * inv[v];
}

// ---------------------------------------------------------------------------
extern "C" void kernel_launch(void* const* d_in, const int* in_sizes, int n_in,
                              void* d_out, int out_size, void* d_ws, size_t ws_size,
                              hipStream_t stream)
{
  const float* seq  = (const float*)d_in[0];
  const float* pos  = (const float*)d_in[1];
  const float* tim  = (const float*)d_in[2];
  const float* Wq   = (const float*)d_in[3];
  const float* Wk   = (const float*)d_in[4];
  const float* mask = (const float*)d_in[5];
  const int*   caus = (const int*)d_in[6];
  float* out = (float*)d_out;

  // workspace: Qh (8MB) | Kh (8MB) | Vt (8MB), all f16
  _Float16* Qh = (_Float16*)d_ws;
  _Float16* Kh = Qh + (size_t)BB * SS * DKK;
  _Float16* Vt = Kh + (size_t)BB * SS * DKK;

  proj_kernel<<<BB * SS, 256, 0, stream>>>(seq, pos, tim, Wq, Wk, Qh, Kh);
  transpose_kernel<<<BB * (SS / 16) * (FF / 16), 256, 0, stream>>>(seq, Vt);
  attn_kernel<<<(BB * SS / 16) / 4, 128, 0, stream>>>(Qh, Kh, Vt, mask, caus, out);
}
